// WhitneyWedgeL2Projector_1194000908532
// MI455X (gfx1250) — matrix operand order repaired and checked by reading, not compiled, not run on hardware
//
#include <hip/hip_runtime.h>

// ---------------------------------------------------------------------------
// WhitneyWedgeL2Projector for MI455X (gfx1250, wave32)
//
//   m_at[t,w,c] = m_par[t,w] * sum_{u,v} triple[t,u,v,w] * ka[t,u,c] * la[t,v,c]
//   out[m_idx[t,w], c] += m_at[t,w,c]          (atomic f32, == segment_sum)
//
// Per wave: groups of 4 tets.  triple_prod (576 B/tet) streamed by TDM into a
// double-buffered LDS tile.  Contraction done as WMMA f16->f32:
//   A (16x32 f16): rows = 2 tets x 8 channels, K = j (uv-pair index, padded 36->64)
//   B (32x16 f16): cols = 4 tets x 4 faces w,   K = j
//   D[m,n] valid only where tet(m)==tet(n)  -> diagonal blocks read, rest ignored.
// Two A-"series" (tets {0,1} and {2,3}) x 2 K-chunks = 4 v_wmma per 4 tets.
// ---------------------------------------------------------------------------

typedef __fp16   fp16x2 __attribute__((ext_vector_type(2)));
typedef _Float16 v8h  __attribute__((ext_vector_type(8)));
typedef _Float16 v16h __attribute__((ext_vector_type(16)));
typedef float    v8f  __attribute__((ext_vector_type(8)));
typedef unsigned int v4u __attribute__((ext_vector_type(4)));
typedef int      v4i  __attribute__((ext_vector_type(4)));
typedef int      v8i  __attribute__((ext_vector_type(8)));

#define WAVES_PER_BLOCK 4
#define BLOCK_THREADS   128
#define KFACES 6
#define MFACES 4
#define CHN    8
#define TRIPLE_PER_TET 144                      // 6*6*4
#define GROUP_TETS 4
#define GROUP_TRIPLE_FLOATS (GROUP_TETS * TRIPLE_PER_TET)   // 576
#define JPAD 64                                 // K extent padded to 2 WMMA chunks

__device__ __forceinline__ unsigned pack_h2(float a, float b) {
    fp16x2 h = __builtin_amdgcn_cvt_pkrtz(a, b);   // one v_cvt_pk_f16_f32
    return __builtin_bit_cast(unsigned, h);
}

#if __has_builtin(__builtin_amdgcn_tensor_load_to_lds)
#define HAVE_TDM 1
#endif

#ifdef HAVE_TDM
// 1-D TDM copy: 2304 bytes (288 x 8B elements) global -> LDS, one descriptor.
__device__ __forceinline__ void tdm_load_group(const float* gsrc, unsigned lds_byte_off) {
    unsigned long long ga = (unsigned long long)(uintptr_t)gsrc;
    unsigned w1 = (unsigned)__builtin_amdgcn_readfirstlane((int)lds_byte_off);
    unsigned w2 = (unsigned)__builtin_amdgcn_readfirstlane((int)(unsigned)ga);
    unsigned w3 = ((unsigned)__builtin_amdgcn_readfirstlane((int)(unsigned)(ga >> 32)) & 0x1FFFFFFu)
                  | (2u << 30);                                  // type=2 ("image")
    v4u g0 = { 1u /*count=1*/, w1 /*lds_addr*/, w2 /*gaddr lo*/, w3 /*gaddr hi|type*/ };
    v8i g1 = { (int)(3u << 16),      // data_size = 3 (8B elements)
               (int)(288u << 16),    // tensor_dim0[15:0] = 288
               (int)(1u  << 16),     // tensor_dim0 hi = 0, tensor_dim1 lo = 1
               (int)(288u << 16),    // tensor_dim1 hi = 0, tile_dim0 = 288
               (int)1,               // tile_dim1 = 1, tile_dim2 = 0
               (int)288,             // tensor_dim0_stride lo = 288
               0, 0 };
    v4i z4 = { 0, 0, 0, 0 };
    v8i z8 = { 0, 0, 0, 0, 0, 0, 0, 0 };
    // 6-arg form on this toolchain: (g0, g1, g2, g3, <aux>, cpol)
    __builtin_amdgcn_tensor_load_to_lds(g0, g1, z4, z4, z8, 0);
}
#endif

__global__ void __launch_bounds__(BLOCK_THREADS)
wedge_main(const float* __restrict__ k_co, const float* __restrict__ l_co,
           const int*   __restrict__ k_idx, const float* __restrict__ k_par,
           const int*   __restrict__ l_idx, const float* __restrict__ l_par,
           const int*   __restrict__ m_idx, const float* __restrict__ m_par,
           const float* __restrict__ triple, float* __restrict__ out,
           int nGroups)
{
    __shared__ __align__(16) float    sRaw[WAVES_PER_BLOCK][2][GROUP_TRIPLE_FLOATS]; // TDM dst
    __shared__ __align__(16) _Float16 sKL [WAVES_PER_BLOCK][GROUP_TETS * CHN    * JPAD]; // [tt*8+c][j]
    __shared__ __align__(16) _Float16 sTP [WAVES_PER_BLOCK][GROUP_TETS * MFACES * JPAD]; // [tt*4+w][j]

    const int lane = threadIdx.x & 31;
    const int wv   = threadIdx.x >> 5;
    const int hi   = lane >> 4;                      // lane half
    const int wavesTotal = gridDim.x * WAVES_PER_BLOCK;
    const int wgid = blockIdx.x * WAVES_PER_BLOCK + wv;

    // Zero the K padding (j = 36..63) once; j<36 is rewritten every group.
    {
        unsigned* p = (unsigned*)&sKL[wv][lane * JPAD + 36];     // 32 rows, 1 per lane
        #pragma unroll
        for (int i = 0; i < 14; ++i) p[i] = 0;
        int row = lane >> 1, h = lane & 1;                       // 16 rows, 2 lanes each
        unsigned* q = (unsigned*)&sTP[wv][row * JPAD + 36 + 14 * h];
        #pragma unroll
        for (int i = 0; i < 7; ++i) q[i] = 0;
    }

    int g = wgid;
    int buf = 0;
#ifdef HAVE_TDM
    if (g < nGroups)
        tdm_load_group(triple + (size_t)g * GROUP_TRIPLE_FLOATS,
                       (unsigned)(uintptr_t)&sRaw[wv][0][0]);
#endif

    for (; g < nGroups; g += wavesTotal) {
        const int gn = g + wavesTotal;
#ifdef HAVE_TDM
        if (gn < nGroups) {
            tdm_load_group(triple + (size_t)gn * GROUP_TRIPLE_FLOATS,
                           (unsigned)(uintptr_t)&sRaw[wv][buf ^ 1][0]);
            __builtin_amdgcn_s_wait_tensorcnt((short)1);   // our older TDM op is done
        } else {
            __builtin_amdgcn_s_wait_tensorcnt((short)0);
        }
        asm volatile("" ::: "memory");
#else
        { // fallback: coalesced wave copy
            const float* src = triple + (size_t)g * GROUP_TRIPLE_FLOATS;
            #pragma unroll
            for (int i = 0; i < GROUP_TRIPLE_FLOATS / 32; ++i)
                sRaw[wv][buf][lane + 32 * i] = src[lane + 32 * i];
        }
#endif

        // ---- stage A: gather cochains with parity, build KL[tt][c][j] in f16 ----
        {
            const int tt = lane >> 3, c = lane & 7;              // 4 tets x 8 channels
            const long t = (long)g * GROUP_TETS + tt;
            float ka[KFACES], la[KFACES];
            #pragma unroll
            for (int u = 0; u < KFACES; ++u) {
                ka[u] = k_par[t * KFACES + u] * k_co[(size_t)k_idx[t * KFACES + u] * CHN + c];
                la[u] = l_par[t * KFACES + u] * l_co[(size_t)l_idx[t * KFACES + u] * CHN + c];
            }
            unsigned* dst = (unsigned*)&sKL[wv][(tt * CHN + c) * JPAD];
            #pragma unroll
            for (int jj = 0; jj < 18; ++jj) {
                const int j0 = 2 * jj, j1 = 2 * jj + 1;          // j = 6u + v
                dst[jj] = pack_h2(ka[j0 / 6] * la[j0 % 6], ka[j1 / 6] * la[j1 % 6]);
            }
        }
        // ---- stage B: transpose triple -> TP[tt][w][j] in f16 (j contiguous) ----
        {
            const int tt = lane >> 3, w = (lane >> 1) & 3, h = lane & 1;
            const float* r = &sRaw[wv][buf][tt * TRIPLE_PER_TET];
            unsigned* dst = (unsigned*)&sTP[wv][(tt * MFACES + w) * JPAD + 18 * h];
            #pragma unroll
            for (int jj = 0; jj < 9; ++jj) {
                const int j = 18 * h + 2 * jj;
                dst[jj] = pack_h2(r[j * 4 + w], r[(j + 1) * 4 + w]);
            }
        }

        // ---- stage C: 4 x v_wmma_f32_16x16x32_f16 ----
        v8f acc0 = {}, acc1 = {};
        const int m = lane & 15;
        #pragma unroll
        for (int kk = 0; kk < 2; ++kk) {
            // B: col n = m = tt_n*4+w ; per-lane halves = K = 32kk + 16*hi + 0..15
            const _Float16* bp = &sTP[wv][m * JPAD + 32 * kk + 16 * hi];
            v8h b_lo = *(const v8h*)(bp);
            v8h b_hi = *(const v8h*)(bp + 8);
            v16h B = __builtin_shufflevector(b_lo, b_hi,
                      0,1,2,3,4,5,6,7,8,9,10,11,12,13,14,15);
            #pragma unroll
            for (int s = 0; s < 2; ++s) {
                // A: row m -> tet 2s + (m>>3), channel m&7 ; halves K = 32kk+16hi+0..15
                const int tt_m = 2 * s + (m >> 3), c = m & 7;
                const _Float16* ap = &sKL[wv][(tt_m * CHN + c) * JPAD + 32 * kk + 16 * hi];
                v8h a_lo = *(const v8h*)(ap);
                v8h a_hi = *(const v8h*)(ap + 8);
                v16h A = __builtin_shufflevector(a_lo, a_hi,
                          0,1,2,3,4,5,6,7,8,9,10,11,12,13,14,15);
                v8f& acc = s ? acc1 : acc0;
                acc = __builtin_amdgcn_wmma_f32_16x16x32_f16(
                          false, A, false, B, (short)0, acc, false, false);
            }
        }

        // ---- stage D: pick diagonal block, apply m-parity, atomic scatter ----
        {
            const int n = m, tt_n = n >> 2, w = n & 3;
            if ((tt_n & 1) == hi) {                     // this lane holds tet tt_n's rows
                const v8f d = (tt_n >> 1) ? acc1 : acc0;
                const long t = (long)g * GROUP_TETS + tt_n;
                const int   mi = m_idx[t * MFACES + w];
                const float mp = m_par[t * MFACES + w];
                float* dst = out + (size_t)mi * CHN;
                #pragma unroll
                for (int cc = 0; cc < CHN; ++cc)
                    atomicAdd(dst + cc, d[cc] * mp);
            }
        }
        buf ^= 1;
    }
}

// Tail tets (T % 4 != 0) — trivially small, plain VALU path.
__global__ void wedge_tail(const float* __restrict__ k_co, const float* __restrict__ l_co,
                           const int* __restrict__ k_idx, const float* __restrict__ k_par,
                           const int* __restrict__ l_idx, const float* __restrict__ l_par,
                           const int* __restrict__ m_idx, const float* __restrict__ m_par,
                           const float* __restrict__ triple, float* __restrict__ out,
                           int tStart, int tCount)
{
    int tid = blockIdx.x * blockDim.x + threadIdx.x;
    if (tid >= tCount * MFACES * CHN) return;
    const int c = tid & 7, w = (tid >> 3) & 3;
    const long t = tStart + (tid >> 5);
    float acc = 0.f;
    for (int u = 0; u < KFACES; ++u) {
        float ka = k_par[t * KFACES + u] * k_co[(size_t)k_idx[t * KFACES + u] * CHN + c];
        for (int v = 0; v < KFACES; ++v) {
            float la = l_par[t * KFACES + v] * l_co[(size_t)l_idx[t * KFACES + v] * CHN + c];
            acc += triple[t * TRIPLE_PER_TET + (u * 6 + v) * 4 + w] * ka * la;
        }
    }
    atomicAdd(out + (size_t)m_idx[t * MFACES + w] * CHN + c, acc * m_par[t * MFACES + w]);
}

extern "C" void kernel_launch(void* const* d_in, const int* in_sizes, int n_in,
                              void* d_out, int out_size, void* d_ws, size_t ws_size,
                              hipStream_t stream)
{
    (void)n_in; (void)d_ws; (void)ws_size;
    const float* k_co  = (const float*)d_in[0];
    const float* l_co  = (const float*)d_in[1];
    const int*   k_idx = (const int*)  d_in[2];   // jax default: int32
    const float* k_par = (const float*)d_in[3];
    const int*   l_idx = (const int*)  d_in[4];
    const float* l_par = (const float*)d_in[5];
    const int*   m_idx = (const int*)  d_in[6];
    const float* m_par = (const float*)d_in[7];
    const float* triple = (const float*)d_in[8];
    float* out = (float*)d_out;

    const int T = in_sizes[8] / TRIPLE_PER_TET;
    const int nGroups = T / GROUP_TETS;
    const int tail = T - nGroups * GROUP_TETS;

    (void)hipMemsetAsync(d_out, 0, (size_t)out_size * sizeof(float), stream);

    if (nGroups > 0) {
        int grid = 2048;                    // 8192 waves -> ~30 groups each
        if (grid > nGroups) grid = nGroups;
        wedge_main<<<grid, BLOCK_THREADS, 0, stream>>>(
            k_co, l_co, k_idx, k_par, l_idx, l_par, m_idx, m_par, triple, out, nGroups);
    }
    if (tail > 0) {
        int thr = tail * MFACES * CHN;
        wedge_tail<<<(thr + 127) / 128, 128, 0, stream>>>(
            k_co, l_co, k_idx, k_par, l_idx, l_par, m_idx, m_par, triple, out,
            nGroups * GROUP_TETS, tail);
    }
}
